// GeneDrugRGCN_39152921870699
// MI455X (gfx1250) — compile-verified, matrix operand order checked by reading, not run on hardware
//
#include <hip/hip_runtime.h>
#include <hip/hip_bf16.h>

typedef __attribute__((ext_vector_type(16))) _Float16 v16h;
typedef __attribute__((ext_vector_type(8)))  float    v8f;

#define NG 100000
#define NDRUG 50000
#define NCLS 2000
#define NFAM 5000
#define ECG 2000000
#define ECP 100000
#define EGF 200000
#define NPAIR 100000

// Guaranteed hardware global_atomic_add_f32 (no CAS-loop fallback)
__device__ __forceinline__ void atomic_add_f32(float* p, float v) {
    unsafeAtomicAdd(p, v);
}

// ---------------- utility kernels ----------------

__global__ void zero_kernel(float* __restrict__ p, long n4) {
    long gid = (long)blockIdx.x * blockDim.x + threadIdx.x;
    if (gid < n4) ((float4*)p)[gid] = make_float4(0.f, 0.f, 0.f, 0.f);
}

__global__ void count_kernel(const int* __restrict__ dst, float* __restrict__ cnt, int E) {
    int e = blockIdx.x * blockDim.x + threadIdx.x;
    if (e < E) atomic_add_f32(&cnt[dst[e]], 1.0f);
}

// 16 threads per edge, float4 each -> 64 floats/row
__global__ void scatter_kernel(const float* __restrict__ xsrc, const int* __restrict__ src,
                               const int* __restrict__ dst, float* __restrict__ agg, int E) {
    long gid = (long)blockIdx.x * blockDim.x + threadIdx.x;
    int e = (int)(gid >> 4);
    if (e >= E) return;
    int c = ((int)gid & 15) * 4;
    int s = src[e], d = dst[e];
    float4 v = *(const float4*)(xsrc + (size_t)s * 64 + c);
    float* a = agg + (size_t)d * 64 + c;
    atomic_add_f32(a + 0, v.x); atomic_add_f32(a + 1, v.y);
    atomic_add_f32(a + 2, v.z); atomic_add_f32(a + 3, v.w);
}

// WT[n*K + k] = (half)(Wa[k*64+n] (+ Wb[k*64+n]))   (W row-major [K,64])
__global__ void wt_kernel(const float* __restrict__ Wa, const float* __restrict__ Wb,
                          _Float16* __restrict__ WT, int K) {
    int gid = blockIdx.x * blockDim.x + threadIdx.x;
    if (gid >= 64 * K) return;
    int n = gid / K, k = gid % K;
    float v = Wa[k * 64 + n];
    if (Wb) v += Wb[k * 64 + n];
    WT[gid] = (_Float16)v;
}

__global__ void bias_kernel(const float* __restrict__ a, const float* __restrict__ b,
                            float* __restrict__ o) {
    int j = threadIdx.x;
    if (j < 64) o[j] = a[j] + (b ? b[j] : 0.f);
}

// ---------------- WMMA fragment helpers (16-bit A layout, ISA 7.12.2) ----------------

__device__ __forceinline__ v16h make_a_frag(const float* __restrict__ rowbase, float scale,
                                            int kstep, int kh) {
    const float* p = rowbase + kstep * 32 + kh * 8;
    v16h a;
#pragma unroll
    for (int j = 0; j < 8; ++j) {
        a[j]     = (_Float16)(p[j] * scale);
        a[j + 8] = (_Float16)(p[j + 16] * scale);
    }
    return a;
}

__device__ __forceinline__ v16h make_b_frag(const _Float16* __restrict__ WT, int ldk,
                                            int n, int kstep, int kh) {
    const _Float16* q = WT + (size_t)n * ldk + kstep * 32 + kh * 8;
    v16h b;
#pragma unroll
    for (int j = 0; j < 8; ++j) {
        b[j]     = q[j];
        b[j + 8] = q[j + 16];
    }
    return b;
}

// ---------------- fused SAGE layer kernel: out = meanA@WmA (+ meanB@WmB) + X@Wr + bias ----------------

__global__ __launch_bounds__(256) void sage_gemm_kernel(
    float* __restrict__ out,
    const float* __restrict__ X,    const _Float16* __restrict__ WrT,
    const float* __restrict__ aggA, const float* __restrict__ cntA, const _Float16* __restrict__ WmAT,
    const float* __restrict__ aggB, const float* __restrict__ cntB, const _Float16* __restrict__ WmBT,
    const float* __restrict__ bias, int M, int relu)
{
    int wave = threadIdx.x >> 5;
    int lane = threadIdx.x & 31;
    int tile_m = (blockIdx.x * 8 + wave) * 16;
    if (tile_m >= M) return;                 // wave-uniform exit, EXEC stays full

    int m  = lane & 15;
    int kh = lane >> 4;
    int row = tile_m + m; if (row >= M) row = M - 1;

    v8f acc[4] = {};

    // root term: X @ Wr
#pragma unroll
    for (int ks = 0; ks < 2; ++ks) {
        v16h a = make_a_frag(X + (size_t)row * 64, 1.0f, ks, kh);
#pragma unroll
        for (int nt = 0; nt < 4; ++nt) {
            v16h b = make_b_frag(WrT, 64, nt * 16 + m, ks, kh);
            acc[nt] = __builtin_amdgcn_wmma_f32_16x16x32_f16(false, a, false, b,
                                                             (short)0, acc[nt], false, false);
        }
    }
    // relation A: mean_A @ WmA
    {
        float s = 1.0f / fmaxf(cntA[row], 1.0f);
#pragma unroll
        for (int ks = 0; ks < 2; ++ks) {
            v16h a = make_a_frag(aggA + (size_t)row * 64, s, ks, kh);
#pragma unroll
            for (int nt = 0; nt < 4; ++nt) {
                v16h b = make_b_frag(WmAT, 64, nt * 16 + m, ks, kh);
                acc[nt] = __builtin_amdgcn_wmma_f32_16x16x32_f16(false, a, false, b,
                                                                 (short)0, acc[nt], false, false);
            }
        }
    }
    // optional relation B
    if (aggB) {
        float s = 1.0f / fmaxf(cntB[row], 1.0f);
#pragma unroll
        for (int ks = 0; ks < 2; ++ks) {
            v16h a = make_a_frag(aggB + (size_t)row * 64, s, ks, kh);
#pragma unroll
            for (int nt = 0; nt < 4; ++nt) {
                v16h b = make_b_frag(WmBT, 64, nt * 16 + m, ks, kh);
                acc[nt] = __builtin_amdgcn_wmma_f32_16x16x32_f16(false, a, false, b,
                                                                 (short)0, acc[nt], false, false);
            }
        }
    }

    // epilogue (C layout: lane n = col, vgpr v = row v + 8*kh)
    int orow_base = tile_m + kh * 8;
#pragma unroll
    for (int nt = 0; nt < 4; ++nt) {
        int col = nt * 16 + m;
        float bv = bias[col];
#pragma unroll
        for (int v = 0; v < 8; ++v) {
            int orow = orow_base + v;
            if (orow < M) {
                float x = acc[nt][v] + bv;
                if (relu) x = fmaxf(x, 0.f);
                out[(size_t)orow * 64 + col] = x;
            }
        }
    }
}

// ---------------- pair MLP stage 1: h1 = relu(concat(comp2[pd], gene2[pg]) @ Wc1 + bc1) ----------------

__global__ __launch_bounds__(256) void mlp1_kernel(
    float* __restrict__ h1,
    const float* __restrict__ comp2, const float* __restrict__ gene2,
    const int* __restrict__ pd, const int* __restrict__ pg,
    const _Float16* __restrict__ Wc1T, const float* __restrict__ bc1, int P)
{
    int wave = threadIdx.x >> 5;
    int lane = threadIdx.x & 31;
    int tile_m = (blockIdx.x * 8 + wave) * 16;
    if (tile_m >= P) return;

    int m  = lane & 15;
    int kh = lane >> 4;
    int row = tile_m + m; if (row >= P) row = P - 1;
    const float* src0 = comp2 + (size_t)pd[row] * 64;
    const float* src1 = gene2 + (size_t)pg[row] * 64;

    v8f acc[4] = {};
#pragma unroll
    for (int ks = 0; ks < 4; ++ks) {
        const float* base = (ks < 2) ? src0 : src1;
        v16h a = make_a_frag(base, 1.0f, ks & 1, kh);
#pragma unroll
        for (int nt = 0; nt < 4; ++nt) {
            v16h b = make_b_frag(Wc1T, 128, nt * 16 + m, ks, kh);
            acc[nt] = __builtin_amdgcn_wmma_f32_16x16x32_f16(false, a, false, b,
                                                             (short)0, acc[nt], false, false);
        }
    }

    int orow_base = tile_m + kh * 8;
#pragma unroll
    for (int nt = 0; nt < 4; ++nt) {
        int col = nt * 16 + m;
        float bv = bc1[col];
#pragma unroll
        for (int v = 0; v < 8; ++v) {
            int orow = orow_base + v;
            if (orow < P) h1[(size_t)orow * 64 + col] = fmaxf(acc[nt][v] + bv, 0.f);
        }
    }
}

// ---------------- pair MLP stage 2: out[p,4] = h1[p] @ Wc2 + bc2 ----------------

__global__ void mlp2_kernel(float* __restrict__ out, const float* __restrict__ h1,
                            const float* __restrict__ Wc2, const float* __restrict__ bc2, int P) {
    int gid = blockIdx.x * blockDim.x + threadIdx.x;
    if (gid >= P * 4) return;
    int p = gid >> 2, j = gid & 3;
    const float* h = h1 + (size_t)p * 64;
    float s = bc2[j];
#pragma unroll
    for (int k = 0; k < 64; ++k) s += h[k] * Wc2[k * 4 + j];
    out[gid] = s;
}

// ---------------- host launch ----------------

extern "C" void kernel_launch(void* const* d_in, const int* in_sizes, int n_in,
                              void* d_out, int out_size, void* d_ws, size_t ws_size,
                              hipStream_t stream) {
    const float* gene_emb   = (const float*)d_in[0];
    const float* drug_emb   = (const float*)d_in[1];
    const float* class_emb  = (const float*)d_in[2];
    const float* family_emb = (const float*)d_in[3];
    const float* W_msg      = (const float*)d_in[4];   // [2,6,64,64]
    const float* b_msg      = (const float*)d_in[5];   // [2,6,64]
    const float* W_root     = (const float*)d_in[6];   // [2,6,64,64]
    const float* Wc1        = (const float*)d_in[7];   // [128,64]
    const float* bc1        = (const float*)d_in[8];
    const float* Wc2        = (const float*)d_in[9];   // [64,4]
    const float* bc2        = (const float*)d_in[10];
    const int* e_cg_src = (const int*)d_in[11]; const int* e_cg_dst = (const int*)d_in[12];
    const int* e_gc_src = (const int*)d_in[13]; const int* e_gc_dst = (const int*)d_in[14];
    const int* e_cp_src = (const int*)d_in[15]; const int* e_cp_dst = (const int*)d_in[16];
    const int* e_pc_src = (const int*)d_in[17]; const int* e_pc_dst = (const int*)d_in[18];
    const int* e_gf_src = (const int*)d_in[19]; const int* e_gf_dst = (const int*)d_in[20];
    const int* e_fg_src = (const int*)d_in[21]; const int* e_fg_dst = (const int*)d_in[22];
    const int* pair_drug = (const int*)d_in[23];
    const int* pair_gene = (const int*)d_in[24];
    float* out = (float*)d_out;

    // workspace bump allocator (256B aligned)
    char* ws = (char*)d_ws;
    size_t off = 0;
    auto allocF = [&](size_t n) -> float* {
        float* p = (float*)(ws + off); off += ((n * 4 + 255) / 256) * 256; return p;
    };
    auto allocH = [&](size_t n) -> _Float16* {
        _Float16* p = (_Float16*)(ws + off); off += ((n * 2 + 255) / 256) * 256; return p;
    };

    float* cnt_r0 = allocF(NG);    float* cnt_r1 = allocF(NDRUG);
    float* cnt_r2 = allocF(NCLS);  float* cnt_r3 = allocF(NDRUG);
    float* cnt_r4 = allocF(NFAM);  float* cnt_r5 = allocF(NG);

    float* agg_g0 = allocF((size_t)NG * 64);    float* agg_g5 = allocF((size_t)NG * 64);
    float* agg_c1 = allocF((size_t)NDRUG * 64); float* agg_c3 = allocF((size_t)NDRUG * 64);
    float* agg_p2 = allocF((size_t)NCLS * 64);  float* agg_f4 = allocF((size_t)NFAM * 64);

    float* x1_g = allocF((size_t)NG * 64);    float* x1_c = allocF((size_t)NDRUG * 64);
    float* x1_p = allocF((size_t)NCLS * 64);  float* x1_f = allocF((size_t)NFAM * 64);
    float* x2_g = allocF((size_t)NG * 64);    float* x2_c = allocF((size_t)NDRUG * 64);

    _Float16* WmT1[6];
    for (int r = 0; r < 6; ++r) WmT1[r] = allocH(4096);
    _Float16* WrT1_g = allocH(4096); _Float16* WrT1_c = allocH(4096);
    _Float16* WrT1_p = allocH(4096); _Float16* WrT1_f = allocH(4096);
    _Float16* WmT2_0 = allocH(4096); _Float16* WmT2_1 = allocH(4096);
    _Float16* WmT2_3 = allocH(4096); _Float16* WmT2_5 = allocH(4096);
    _Float16* WrT2_g = allocH(4096); _Float16* WrT2_c = allocH(4096);
    _Float16* Wc1T   = allocH(64 * 128);

    float* bz1_g = allocF(64); float* bz1_c = allocF(64);
    float* bz1_p = allocF(64); float* bz1_f = allocF(64);
    float* bz2_g = allocF(64); float* bz2_c = allocF(64);

    float* h1 = agg_g0;  // alias: aggs are dead by MLP time (NG*64 == NPAIR*64)

    auto cdiv = [](long a, long b) { return (int)((a + b - 1) / b); };
    const int B = 256;

    // --- counts (edge lists identical across layers: compute once per call) ---
    zero_kernel<<<cdiv(NG / 4, B), B, 0, stream>>>(cnt_r0, NG / 4);
    zero_kernel<<<cdiv(NDRUG / 4, B), B, 0, stream>>>(cnt_r1, NDRUG / 4);
    zero_kernel<<<cdiv(NCLS / 4, B), B, 0, stream>>>(cnt_r2, NCLS / 4);
    zero_kernel<<<cdiv(NDRUG / 4, B), B, 0, stream>>>(cnt_r3, NDRUG / 4);
    zero_kernel<<<cdiv(NFAM / 4, B), B, 0, stream>>>(cnt_r4, NFAM / 4);
    zero_kernel<<<cdiv(NG / 4, B), B, 0, stream>>>(cnt_r5, NG / 4);
    count_kernel<<<cdiv(ECG, B), B, 0, stream>>>(e_cg_dst, cnt_r0, ECG);
    count_kernel<<<cdiv(ECG, B), B, 0, stream>>>(e_gc_dst, cnt_r1, ECG);
    count_kernel<<<cdiv(ECP, B), B, 0, stream>>>(e_cp_dst, cnt_r2, ECP);
    count_kernel<<<cdiv(ECP, B), B, 0, stream>>>(e_pc_dst, cnt_r3, ECP);
    count_kernel<<<cdiv(EGF, B), B, 0, stream>>>(e_gf_dst, cnt_r4, EGF);
    count_kernel<<<cdiv(EGF, B), B, 0, stream>>>(e_fg_dst, cnt_r5, EGF);

    // --- weight prep: transposed f16; HeteroConv-sum folded into combined root W / bias ---
    const int WG = cdiv(64 * 64, B);
    for (int r = 0; r < 6; ++r)
        wt_kernel<<<WG, B, 0, stream>>>(W_msg + (size_t)r * 4096, nullptr, WmT1[r], 64);
    wt_kernel<<<WG, B, 0, stream>>>(W_root + 0 * 4096, W_root + 5 * 4096, WrT1_g, 64);
    wt_kernel<<<WG, B, 0, stream>>>(W_root + 1 * 4096, W_root + 3 * 4096, WrT1_c, 64);
    wt_kernel<<<WG, B, 0, stream>>>(W_root + 2 * 4096, nullptr, WrT1_p, 64);
    wt_kernel<<<WG, B, 0, stream>>>(W_root + 4 * 4096, nullptr, WrT1_f, 64);
    const float* Wm2 = W_msg + 6 * 4096;
    const float* Wr2 = W_root + 6 * 4096;
    wt_kernel<<<WG, B, 0, stream>>>(Wm2 + 0 * 4096, nullptr, WmT2_0, 64);
    wt_kernel<<<WG, B, 0, stream>>>(Wm2 + 1 * 4096, nullptr, WmT2_1, 64);
    wt_kernel<<<WG, B, 0, stream>>>(Wm2 + 3 * 4096, nullptr, WmT2_3, 64);
    wt_kernel<<<WG, B, 0, stream>>>(Wm2 + 5 * 4096, nullptr, WmT2_5, 64);
    wt_kernel<<<WG, B, 0, stream>>>(Wr2 + 0 * 4096, Wr2 + 5 * 4096, WrT2_g, 64);
    wt_kernel<<<WG, B, 0, stream>>>(Wr2 + 1 * 4096, Wr2 + 3 * 4096, WrT2_c, 64);
    wt_kernel<<<cdiv(64 * 128, B), B, 0, stream>>>(Wc1, nullptr, Wc1T, 128);

    const float* bm2 = b_msg + 6 * 64;
    bias_kernel<<<1, 64, 0, stream>>>(b_msg + 0 * 64, b_msg + 5 * 64, bz1_g);
    bias_kernel<<<1, 64, 0, stream>>>(b_msg + 1 * 64, b_msg + 3 * 64, bz1_c);
    bias_kernel<<<1, 64, 0, stream>>>(b_msg + 2 * 64, nullptr, bz1_p);
    bias_kernel<<<1, 64, 0, stream>>>(b_msg + 4 * 64, nullptr, bz1_f);
    bias_kernel<<<1, 64, 0, stream>>>(bm2 + 0 * 64, bm2 + 5 * 64, bz2_g);
    bias_kernel<<<1, 64, 0, stream>>>(bm2 + 1 * 64, bm2 + 3 * 64, bz2_c);

    // --- layer 1: all 6 relations ---
    zero_kernel<<<cdiv((long)NG * 16, B), B, 0, stream>>>(agg_g0, (long)NG * 16);
    zero_kernel<<<cdiv((long)NG * 16, B), B, 0, stream>>>(agg_g5, (long)NG * 16);
    zero_kernel<<<cdiv((long)NDRUG * 16, B), B, 0, stream>>>(agg_c1, (long)NDRUG * 16);
    zero_kernel<<<cdiv((long)NDRUG * 16, B), B, 0, stream>>>(agg_c3, (long)NDRUG * 16);
    zero_kernel<<<cdiv((long)NCLS * 16, B), B, 0, stream>>>(agg_p2, (long)NCLS * 16);
    zero_kernel<<<cdiv((long)NFAM * 16, B), B, 0, stream>>>(agg_f4, (long)NFAM * 16);
    scatter_kernel<<<cdiv((long)ECG * 16, B), B, 0, stream>>>(drug_emb, e_cg_src, e_cg_dst, agg_g0, ECG);
    scatter_kernel<<<cdiv((long)ECG * 16, B), B, 0, stream>>>(gene_emb, e_gc_src, e_gc_dst, agg_c1, ECG);
    scatter_kernel<<<cdiv((long)ECP * 16, B), B, 0, stream>>>(drug_emb, e_cp_src, e_cp_dst, agg_p2, ECP);
    scatter_kernel<<<cdiv((long)ECP * 16, B), B, 0, stream>>>(class_emb, e_pc_src, e_pc_dst, agg_c3, ECP);
    scatter_kernel<<<cdiv((long)EGF * 16, B), B, 0, stream>>>(gene_emb, e_gf_src, e_gf_dst, agg_f4, EGF);
    scatter_kernel<<<cdiv((long)EGF * 16, B), B, 0, stream>>>(family_emb, e_fg_src, e_fg_dst, agg_g5, EGF);

    sage_gemm_kernel<<<cdiv(NG, 128), B, 0, stream>>>(x1_g, gene_emb, WrT1_g,
        agg_g0, cnt_r0, WmT1[0], agg_g5, cnt_r5, WmT1[5], bz1_g, NG, 1);
    sage_gemm_kernel<<<cdiv(NDRUG, 128), B, 0, stream>>>(x1_c, drug_emb, WrT1_c,
        agg_c1, cnt_r1, WmT1[1], agg_c3, cnt_r3, WmT1[3], bz1_c, NDRUG, 1);
    sage_gemm_kernel<<<cdiv(NCLS, 128), B, 0, stream>>>(x1_p, class_emb, WrT1_p,
        agg_p2, cnt_r2, WmT1[2], nullptr, nullptr, nullptr, bz1_p, NCLS, 1);
    sage_gemm_kernel<<<cdiv(NFAM, 128), B, 0, stream>>>(x1_f, family_emb, WrT1_f,
        agg_f4, cnt_r4, WmT1[4], nullptr, nullptr, nullptr, bz1_f, NFAM, 1);

    // --- layer 2: only gene/compound outputs are consumed downstream ---
    zero_kernel<<<cdiv((long)NG * 16, B), B, 0, stream>>>(agg_g0, (long)NG * 16);
    zero_kernel<<<cdiv((long)NG * 16, B), B, 0, stream>>>(agg_g5, (long)NG * 16);
    zero_kernel<<<cdiv((long)NDRUG * 16, B), B, 0, stream>>>(agg_c1, (long)NDRUG * 16);
    zero_kernel<<<cdiv((long)NDRUG * 16, B), B, 0, stream>>>(agg_c3, (long)NDRUG * 16);
    scatter_kernel<<<cdiv((long)ECG * 16, B), B, 0, stream>>>(x1_c, e_cg_src, e_cg_dst, agg_g0, ECG);
    scatter_kernel<<<cdiv((long)ECG * 16, B), B, 0, stream>>>(x1_g, e_gc_src, e_gc_dst, agg_c1, ECG);
    scatter_kernel<<<cdiv((long)ECP * 16, B), B, 0, stream>>>(x1_p, e_pc_src, e_pc_dst, agg_c3, ECP);
    scatter_kernel<<<cdiv((long)EGF * 16, B), B, 0, stream>>>(x1_f, e_fg_src, e_fg_dst, agg_g5, EGF);

    sage_gemm_kernel<<<cdiv(NG, 128), B, 0, stream>>>(x2_g, x1_g, WrT2_g,
        agg_g0, cnt_r0, WmT2_0, agg_g5, cnt_r5, WmT2_5, bz2_g, NG, 0);
    sage_gemm_kernel<<<cdiv(NDRUG, 128), B, 0, stream>>>(x2_c, x1_c, WrT2_c,
        agg_c1, cnt_r1, WmT2_1, agg_c3, cnt_r3, WmT2_3, bz2_c, NDRUG, 0);

    // --- pair MLP ---
    mlp1_kernel<<<cdiv(NPAIR, 128), B, 0, stream>>>(h1, x2_c, x2_g, pair_drug, pair_gene,
                                                    Wc1T, bc1, NPAIR);
    mlp2_kernel<<<cdiv(NPAIR * 4, B), B, 0, stream>>>(out, h1, Wc2, bc2, NPAIR);
}
